// MultiHeadAttention_26345329393870
// MI455X (gfx1250) — matrix operand order repaired and checked by reading, not compiled, not run on hardware
//
#include <hip/hip_runtime.h>
#include <hip/hip_bf16.h>

// ---------------------------------------------------------------------------
// Complex MHA (B=2,S=2048,D=1024,H=16,DK=64) decomposed into real bf16 WMMA.
//   real/imag stacked as 8192 GEMM rows; K-imag pre-negated; V pre-transposed.
// Workspace layout (bytes), total 75,497,472 (~72 MB):
//   [0        ,16777216) Xbf  (8192x1024 bf16)  -- reused as attention out O
//   [16777216 ,33554432) Q    [ri][b][h][s][dk] bf16
//   [33554432 ,50331648) K    [ri][b][h][s][dk] bf16 (imag negated)
//   [50331648 ,67108864) Vt   [ri][b][h][dk][s] bf16
//   [67108864 ,75497472) WtQ,WtK,WtV,WtO (each 1024x1024 bf16, [e][d])
// ---------------------------------------------------------------------------

typedef __attribute__((ext_vector_type(16))) __bf16 v16bf;
typedef __attribute__((ext_vector_type(8)))  float  v8f;

union FragU { uint4 q[2]; v16bf v; };

static __device__ __forceinline__ v8f wmma_bf16(v16bf a, v16bf b, v8f c) {
  // D = A(16x32 bf16) * B(32x16 bf16) + C(16x16 f32)
  return __builtin_amdgcn_wmma_f32_16x16x32_bf16(false, a, false, b,
                                                 (short)0, c, false, false);
}

// A-fragment (16x32, bf16), storage row-major [m][k], ld in elements.
// Lane L: m=L&15, half=L>>4; VGPR v: k = (v>>2)*16 + half*8 + 2*(v&3) (+kcol)
// => two contiguous 16B loads per lane.
static __device__ __forceinline__ v16bf load_a16x32(const __bf16* base, int ld, int kcol) {
  const int lane = threadIdx.x & 31, m = lane & 15, half = lane >> 4;
  const char* r = (const char*)(base + (size_t)m * ld + kcol);
  FragU u;
  u.q[0] = *(const uint4*)(r + half * 16);
  u.q[1] = *(const uint4*)(r + 32 + half * 16);
  return u.v;
}

// B-fragment (32x16, bf16) from N-major storage Bt[n][k] (ld in elements).
// Lane L: n=L&15, half=L>>4; VGPR v: k = (v>>2)*16 + half*8 + 2*(v&3) (+krow0)
// => two contiguous 16B loads per lane from row n.
static __device__ __forceinline__ v16bf load_bT32x16(const __bf16* base, int ld, int krow0) {
  const int lane = threadIdx.x & 31, n = lane & 15, half = lane >> 4;
  const char* r = (const char*)(base + (size_t)n * ld + krow0);
  FragU u;
  u.q[0] = *(const uint4*)(r + half * 16);
  u.q[1] = *(const uint4*)(r + 32 + half * 16);
  return u.v;
}

// --------------------------- pack kernels ----------------------------------
__global__ __launch_bounds__(256) void pack_x(const float* __restrict__ xr,
                                              const float* __restrict__ xi,
                                              __bf16* __restrict__ xb) {
  const size_t i = (size_t)blockIdx.x * 256 + threadIdx.x;   // 4,194,304 total
  xb[i]             = (__bf16)xr[i];
  xb[i + 4194304ul] = (__bf16)xi[i];
}

__global__ __launch_bounds__(256) void pack_wT(const float* __restrict__ w,
                                               __bf16* __restrict__ wt) {
  const int i = blockIdx.x * 256 + threadIdx.x;              // 1,048,576 total
  const int e = i >> 10, d = i & 1023;
  wt[i] = (__bf16)w[(size_t)d * 1024 + e];                   // wt[e][d] = w[d][e]
}

// --------------------------- shared GEMM -----------------------------------
// C(8192x1024) = A(8192x1024 bf16) * Wt^T + bias(real rows only).
// mode 0: store Q layout  [ri][b][h][s][dk]
// mode 1: store K layout, negate imag rows (pre-negated Ki)
// mode 2: store Vt layout [ri][b][h][dk][s]
// mode 3: store final complex64 (interleaved float) to d_out
__global__ __launch_bounds__(128) void gemm8192(const __bf16* __restrict__ A,
                                                const __bf16* __restrict__ Bt,
                                                const float* __restrict__ bias,
                                                __bf16* __restrict__ outB,
                                                float* __restrict__ outF,
                                                int mode) {
  const int wave = threadIdx.x >> 5;
  const int lane = threadIdx.x & 31, n0 = lane & 15, half = lane >> 4;
  const int mrow = blockIdx.x * 32;
  const int ncol = (blockIdx.y * 4 + wave) * 64;

  v8f acc[2][4];
#pragma unroll
  for (int i = 0; i < 2; ++i)
#pragma unroll
    for (int t = 0; t < 4; ++t) acc[i][t] = v8f{};

  for (int kc = 0; kc < 1024; kc += 32) {
    v16bf a0 = load_a16x32(A + (size_t)mrow * 1024, 1024, kc);
    v16bf a1 = load_a16x32(A + (size_t)(mrow + 16) * 1024, 1024, kc);
#pragma unroll
    for (int t = 0; t < 4; ++t) {
      v16bf bf = load_bT32x16(Bt + (size_t)(ncol + t * 16) * 1024, 1024, kc);
      acc[0][t] = wmma_bf16(a0, bf, acc[0][t]);
      acc[1][t] = wmma_bf16(a1, bf, acc[1][t]);
    }
  }

#pragma unroll
  for (int mt = 0; mt < 2; ++mt)
#pragma unroll
    for (int t = 0; t < 4; ++t)
#pragma unroll
      for (int v = 0; v < 8; ++v) {
        const int row = mrow + mt * 16 + v + 8 * half;
        const int col = ncol + t * 16 + n0;
        float val = acc[mt][t][v];
        const int ri = row >> 12;          // 0=real rows, 1=imag rows
        const int sb = row & 4095;         // b*S + s
        if (ri == 0) val += bias[col];     // complex cast: bias on real only
        if (mode == 3) {
          outF[2 * ((size_t)sb * 1024 + col) + ri] = val;   // interleaved c64
        } else {
          const int b = sb >> 11, s = sb & 2047;
          const int h = col >> 6, dk = col & 63;
          const size_t head = (size_t)(ri * 2 + b) * 16 + h;
          if (mode == 2) {
            outB[head * 131072 + (size_t)dk * 2048 + s] = (__bf16)val;
          } else {
            if (mode == 1 && ri) val = -val;                // pre-negate Ki
            outB[head * 131072 + (size_t)s * 64 + dk] = (__bf16)val;
          }
        }
      }
}

// --------------------------- flash attention -------------------------------
// One wave = 16 queries of one (b,h), streaming 2048 keys in 32-key chunks.
// scores.real = Qr*Kr^T + Qi*(-Ki)^T, online softmax (base-2), O += P*V.
__global__ __launch_bounds__(128) void attn_fwd(const __bf16* __restrict__ Q,
                                                const __bf16* __restrict__ K,
                                                const __bf16* __restrict__ Vt,
                                                __bf16* __restrict__ O) {
  __shared__ __align__(16) __bf16 pb[4][16][32];   // per-wave P transpose tile

  const int wave = threadIdx.x >> 5, lane = threadIdx.x & 31;
  const int n0 = lane & 15, half = lane >> 4;
  const int b = blockIdx.z, h = blockIdx.y;
  const int qblk = blockIdx.x * 4 + wave;          // 0..127
  const size_t head = ((size_t)b * 16 + h) * 131072;  // S*DK
  const size_t RI = 4194304;                       // B*H*S*DK

  const __bf16* qr = Q + head + (size_t)qblk * 16 * 64;
  const __bf16* qi = qr + RI;
  const v16bf aQr0 = load_a16x32(qr, 64, 0);
  const v16bf aQr1 = load_a16x32(qr, 64, 32);
  const v16bf aQi0 = load_a16x32(qi, 64, 0);
  const v16bf aQi1 = load_a16x32(qi, 64, 32);

  const __bf16* Kr = K + head;
  const __bf16* Ki = Kr + RI;                      // holds -imag(K)
  const __bf16* Vr = Vt + head;
  const __bf16* Vi = Vr + RI;

  v8f oAcc[2][4];
#pragma unroll
  for (int i = 0; i < 2; ++i)
#pragma unroll
    for (int t = 0; t < 4; ++t) oAcc[i][t] = v8f{};
  float rm[8], rl[8];
#pragma unroll
  for (int v = 0; v < 8; ++v) { rm[v] = -3.0e38f; rl[v] = 0.0f; }

  const float KS = 0.125f * 1.44269504088896f;     // 1/sqrt(DK) * log2(e)

  for (int kb = 0; kb < 2048; kb += 32) {
    // ---- scores: two 16-key tiles, 4 chained WMMAs each ----
    v8f s0 = v8f{}, s1 = v8f{};
    const __bf16* kr0 = Kr + (size_t)kb * 64;
    const __bf16* ki0 = Ki + (size_t)kb * 64;
    s0 = wmma_bf16(aQr0, load_bT32x16(kr0, 64, 0), s0);
    s0 = wmma_bf16(aQr1, load_bT32x16(kr0, 64, 32), s0);
    s0 = wmma_bf16(aQi0, load_bT32x16(ki0, 64, 0), s0);
    s0 = wmma_bf16(aQi1, load_bT32x16(ki0, 64, 32), s0);
    const __bf16* kr1 = kr0 + 16 * 64;
    const __bf16* ki1 = ki0 + 16 * 64;
    s1 = wmma_bf16(aQr0, load_bT32x16(kr1, 64, 0), s1);
    s1 = wmma_bf16(aQr1, load_bT32x16(kr1, 64, 32), s1);
    s1 = wmma_bf16(aQi0, load_bT32x16(ki1, 64, 0), s1);
    s1 = wmma_bf16(aQi1, load_bT32x16(ki1, 64, 32), s1);

    // ---- online softmax (rows = v + 8*half, cols across 16-lane group) ----
    float p0[8], p1[8];
#pragma unroll
    for (int v = 0; v < 8; ++v) {
      const float z0 = s0[v] * KS, z1 = s1[v] * KS;
      float mx = fmaxf(z0, z1);
      mx = fmaxf(mx, __shfl_xor(mx, 1, 32));
      mx = fmaxf(mx, __shfl_xor(mx, 2, 32));
      mx = fmaxf(mx, __shfl_xor(mx, 4, 32));
      mx = fmaxf(mx, __shfl_xor(mx, 8, 32));
      const float mn = fmaxf(rm[v], mx);
      const float corr = exp2f(rm[v] - mn);
      p0[v] = exp2f(z0 - mn);
      p1[v] = exp2f(z1 - mn);
      float ps = p0[v] + p1[v];
      ps += __shfl_xor(ps, 1, 32);
      ps += __shfl_xor(ps, 2, 32);
      ps += __shfl_xor(ps, 4, 32);
      ps += __shfl_xor(ps, 8, 32);
      rl[v] = rl[v] * corr + ps;
      rm[v] = mn;
#pragma unroll
      for (int t = 0; t < 4; ++t) { oAcc[0][t][v] *= corr; oAcc[1][t][v] *= corr; }
    }

    // ---- P (C-layout) -> bf16 A-fragment via per-wave LDS tile ----
#pragma unroll
    for (int v = 0; v < 8; ++v) {
      const int row = v + 8 * half;
      pb[wave][row][n0]      = (__bf16)p0[v];
      pb[wave][row][16 + n0] = (__bf16)p1[v];
    }
    asm volatile("" ::: "memory");   // same-wave LDS is in-order; block reorder
    const v16bf pa = load_a16x32(&pb[wave][0][0], 32, 0);

    // ---- O += P * V (Vt is [dk][s], the natural B operand) ----
#pragma unroll
    for (int t = 0; t < 4; ++t) {
      oAcc[0][t] = wmma_bf16(pa, load_bT32x16(Vr + (size_t)t * 16 * 2048, 2048, kb), oAcc[0][t]);
      oAcc[1][t] = wmma_bf16(pa, load_bT32x16(Vi + (size_t)t * 16 * 2048, 2048, kb), oAcc[1][t]);
    }
  }

  // ---- finalize: O /= l, store bf16 rows for the Wo GEMM ----
#pragma unroll
  for (int v = 0; v < 8; ++v) {
    const float inv = 1.0f / rl[v];
    const int q = qblk * 16 + v + 8 * half;
    const size_t rowR = ((size_t)b * 2048 + q) * 1024;
#pragma unroll
    for (int t = 0; t < 4; ++t) {
      const int col = h * 64 + t * 16 + n0;
      O[rowR + col]                  = (__bf16)(oAcc[0][t][v] * inv);
      O[rowR + 4194304ul + col]      = (__bf16)(oAcc[1][t][v] * inv);
    }
  }
}

// ---------------------------------------------------------------------------
extern "C" void kernel_launch(void* const* d_in, const int* in_sizes, int n_in,
                              void* d_out, int out_size, void* d_ws, size_t ws_size,
                              hipStream_t stream) {
  const float* xr = (const float*)d_in[0];
  const float* xi = (const float*)d_in[1];
  const float* Wq = (const float*)d_in[2];
  const float* bq = (const float*)d_in[3];
  const float* Wk = (const float*)d_in[4];
  const float* bk = (const float*)d_in[5];
  const float* Wv = (const float*)d_in[6];
  const float* bv = (const float*)d_in[7];
  const float* Wo = (const float*)d_in[8];
  const float* bo = (const float*)d_in[9];

  char* ws = (char*)d_ws;
  __bf16* Xbf = (__bf16*)(ws);                       // also attention output O
  __bf16* Qb  = (__bf16*)(ws + 16777216ul);
  __bf16* Kb  = (__bf16*)(ws + 33554432ul);
  __bf16* Vtb = (__bf16*)(ws + 50331648ul);
  __bf16* WtQ = (__bf16*)(ws + 67108864ul);
  __bf16* WtK = WtQ + 1048576;
  __bf16* WtV = WtK + 1048576;
  __bf16* WtO = WtV + 1048576;

  // 1) pack activations and transposed-bf16 weights
  pack_x<<<16384, 256, 0, stream>>>(xr, xi, Xbf);
  pack_wT<<<4096, 256, 0, stream>>>(Wq, WtQ);
  pack_wT<<<4096, 256, 0, stream>>>(Wk, WtK);
  pack_wT<<<4096, 256, 0, stream>>>(Wv, WtV);
  pack_wT<<<4096, 256, 0, stream>>>(Wo, WtO);

  // 2) projections (real+imag stacked rows)
  dim3 ggrid(256, 4);
  gemm8192<<<ggrid, 128, 0, stream>>>(Xbf, WtQ, bq, Qb,  nullptr, 0);
  gemm8192<<<ggrid, 128, 0, stream>>>(Xbf, WtK, bk, Kb,  nullptr, 1);
  gemm8192<<<ggrid, 128, 0, stream>>>(Xbf, WtV, bv, Vtb, nullptr, 2);

  // 3) flash attention (writes O into the Xbf region, which is now free)
  attn_fwd<<<dim3(32, 16, 2), 128, 0, stream>>>(Qb, Kb, Vtb, Xbf);

  // 4) output projection -> interleaved complex64 floats
  gemm8192<<<ggrid, 128, 0, stream>>>(Xbf, WtO, bo, nullptr, (float*)d_out, 3);
}
